// GroupedQueryAttention_36043365548527
// MI455X (gfx1250) — compile-verified, matrix operand order checked by reading, not run on hardware
//
#include <hip/hip_runtime.h>
#include <hip/hip_bf16.h>
#include <stdint.h>

// Problem constants (match reference)
#define B_    2
#define S_    2048
#define DIN   2048
#define H_    16
#define KV_   4
#define HD_   128
#define GROUP_ (H_ / KV_)
#define SCALE_ 0.08838834764831845f   // 128^-0.5

typedef __bf16 bf16;
typedef __attribute__((ext_vector_type(16))) __bf16 v16bf;
typedef __attribute__((ext_vector_type(8)))  float  v8f;

// CDNA5 async global->LDS copy (no VGPR staging), guarded so the file still
// compiles on toolchains without the builtin.
#if __has_builtin(__builtin_amdgcn_global_load_async_to_lds_b128) && \
    __has_builtin(__builtin_amdgcn_s_wait_asynccnt)
#define HAVE_ASYNC_LDS 1
#else
#define HAVE_ASYNC_LDS 0
#endif

#if HAVE_ASYNC_LDS
// Builtin prototype (from hipcc diagnostic): first arg is a *non-const*
// pointer to 'int __attribute__((vector_size(16)))' in the device (global)
// address space; LDS arg mirrors that in the shared address space.
typedef int vsi4 __attribute__((vector_size(16)));
typedef __attribute__((address_space(1))) vsi4* as1_v4i;
typedef __attribute__((address_space(3))) vsi4* as3_v4i;

__device__ __forceinline__ void async_copy16(const bf16* g, bf16* l) {
  // flat LDS pointer: addr[31:0] is the LDS offset (aperture layout);
  // global flat VA is identity -> integer round-trip builds the AS pointers.
  __builtin_amdgcn_global_load_async_to_lds_b128(
      (as1_v4i)(uintptr_t)g, (as3_v4i)(uint32_t)(uintptr_t)l, 0, 0);
}
__device__ __forceinline__ void async_wait0() {
  __builtin_amdgcn_s_wait_asynccnt(0);
}
#endif

// ---------- helpers ----------

__device__ __forceinline__ bf16 f2bf(float f) {
  union { float f; uint32_t u; } in;
  in.f = f;
  uint32_t u = in.u;
  uint32_t r = (u + 0x7FFFu + ((u >> 16) & 1u)) >> 16;   // round-to-nearest-even
  union { uint16_t s; bf16 h; } out;
  out.s = (uint16_t)r;
  return out.h;
}

// 16-bit A/B fragment for wmma_f32_16x16x32_bf16.
// Per ISA: lane<16 holds k {0..7, 16..23}; lane>=16 holds k {8..15, 24..31}.
// => two contiguous 16B chunks at elem offsets (8*hi) and (16 + 8*hi).
__device__ __forceinline__ v16bf load_frag(const bf16* base, int row, int ldw,
                                           int kOff, int lane) {
  const int hi = (lane >> 4) & 1;
  union { v16bf v; uint4 q[2]; } f;
  const bf16* p = base + (size_t)row * ldw + kOff + 8 * hi;
  f.q[0] = *reinterpret_cast<const uint4*>(p);
  f.q[1] = *reinterpret_cast<const uint4*>(p + 16);
  return f.v;
}

__device__ __forceinline__ v8f wmma_bf16(v16bf a, v16bf b, v8f c) {
  return __builtin_amdgcn_wmma_f32_16x16x32_bf16(false, a, false, b,
                                                 (short)0, c, false, false);
}

// ---------- kernel 1: fp32 -> bf16 ----------

__global__ void k_f32_to_bf16(const float* __restrict__ in,
                              bf16* __restrict__ out, int n) {
  int i = blockIdx.x * blockDim.x + threadIdx.x;
  if (i < n) out[i] = f2bf(in[i]);
}

// ---------- kernel 2: GEMM  C[M,N] = A[M,K] * W[N,K]^T  (bf16 in, f32 out) ----------
// Block tile 128x128, 256 threads = 8 waves (4x2), each wave a 32x64 tile.
// K-step 32, double-buffered LDS. Next tile is moved global->LDS with
// GLOBAL_LOAD_ASYNC_TO_LDS_B128 (ASYNCcnt) while WMMAs consume the current
// buffer -> zero VGPR staging, no spills; fallback path stages via registers.

#define GT_M 128
#define GT_N 128
#define GT_K 32
#define LDT  40   // 32 + 8 pad (keeps 16B alignment, dodges bank conflicts)

__global__ __launch_bounds__(256, 1)
void k_gemm_bf16(const bf16* __restrict__ A, const bf16* __restrict__ W,
                 float* __restrict__ C, int M, int N, int K) {
  __shared__ __align__(16) bf16 As[2][GT_M * LDT];
  __shared__ __align__(16) bf16 Ws[2][GT_N * LDT];

  const int tid  = threadIdx.x;
  const int lane = tid & 31;
  const int w    = tid >> 5;
  const int wr   = w & 3;        // M sub-tile: rows wr*32
  const int wc   = w >> 2;       // N sub-tile: cols wc*64
  const int mb   = blockIdx.y * GT_M;
  const int nb   = blockIdx.x * GT_N;
  const int l15  = lane & 15;
  const int hi   = (lane >> 4) & 1;

  // cooperative load map: tile = 128 rows x 32 cols = 512 uint4 chunks, 2/thread
  int lrow[2], lcol[2];
  for (int i = 0; i < 2; ++i) {
    int c = tid + i * 256;
    lrow[i] = c >> 2;            // 4 chunks per row
    lcol[i] = (c & 3) * 8;
  }

  v8f acc[2][4] = {};

  // prologue: tile 0 -> buffer 0
#if HAVE_ASYNC_LDS
  for (int i = 0; i < 2; ++i) {
    async_copy16(&A[(size_t)(mb + lrow[i]) * K + lcol[i]],
                 &As[0][lrow[i] * LDT + lcol[i]]);
    async_copy16(&W[(size_t)(nb + lrow[i]) * K + lcol[i]],
                 &Ws[0][lrow[i] * LDT + lcol[i]]);
  }
  async_wait0();
#else
  for (int i = 0; i < 2; ++i) {
    *reinterpret_cast<uint4*>(&As[0][lrow[i] * LDT + lcol[i]]) =
        *reinterpret_cast<const uint4*>(&A[(size_t)(mb + lrow[i]) * K + lcol[i]]);
    *reinterpret_cast<uint4*>(&Ws[0][lrow[i] * LDT + lcol[i]]) =
        *reinterpret_cast<const uint4*>(&W[(size_t)(nb + lrow[i]) * K + lcol[i]]);
  }
#endif
  __syncthreads();

  int buf = 0;
  for (int k0 = 0; k0 < K; k0 += GT_K) {
    const bool haveNext = (k0 + GT_K < K);
#if HAVE_ASYNC_LDS
    if (haveNext) {   // async DMA next tile into the other buffer
      for (int i = 0; i < 2; ++i) {
        async_copy16(&A[(size_t)(mb + lrow[i]) * K + k0 + GT_K + lcol[i]],
                     &As[buf ^ 1][lrow[i] * LDT + lcol[i]]);
        async_copy16(&W[(size_t)(nb + lrow[i]) * K + k0 + GT_K + lcol[i]],
                     &Ws[buf ^ 1][lrow[i] * LDT + lcol[i]]);
      }
      if (k0 + 2 * GT_K < K) {   // pull tile k+2 toward L2 (global_prefetch_b8)
        __builtin_prefetch(&A[(size_t)(mb + lrow[0]) * K + k0 + 2 * GT_K + lcol[0]], 0, 1);
        __builtin_prefetch(&W[(size_t)(nb + lrow[0]) * K + k0 + 2 * GT_K + lcol[0]], 0, 1);
      }
    }
#else
    uint4 an[2], wn[2];
    if (haveNext) {   // fetch next tile to registers while we compute
      for (int i = 0; i < 2; ++i) {
        an[i] = *reinterpret_cast<const uint4*>(
            &A[(size_t)(mb + lrow[i]) * K + k0 + GT_K + lcol[i]]);
        wn[i] = *reinterpret_cast<const uint4*>(
            &W[(size_t)(nb + lrow[i]) * K + k0 + GT_K + lcol[i]]);
      }
      if (k0 + 2 * GT_K < K) {
        __builtin_prefetch(&A[(size_t)(mb + lrow[0]) * K + k0 + 2 * GT_K + lcol[0]], 0, 1);
        __builtin_prefetch(&W[(size_t)(nb + lrow[0]) * K + k0 + 2 * GT_K + lcol[0]], 0, 1);
      }
    }
#endif

    // compute on current buffer: A-frags resident, B-frags streamed one at a
    // time (each feeds 2 WMMAs) to minimize the live fragment set.
    v16bf af0 = load_frag(As[buf], wr * 32 + l15, LDT, 0, lane);
    v16bf af1 = load_frag(As[buf], wr * 32 + 16 + l15, LDT, 0, lane);
    for (int nt = 0; nt < 4; ++nt) {
      v16bf bfr = load_frag(Ws[buf], wc * 64 + nt * 16 + l15, LDT, 0, lane);
      acc[0][nt] = wmma_bf16(af0, bfr, acc[0][nt]);
      acc[1][nt] = wmma_bf16(af1, bfr, acc[1][nt]);
    }

#if HAVE_ASYNC_LDS
    async_wait0();          // our async writes to buf^1 are in LDS
#else
    if (haveNext) {
      for (int i = 0; i < 2; ++i) {
        *reinterpret_cast<uint4*>(&As[buf ^ 1][lrow[i] * LDT + lcol[i]]) = an[i];
        *reinterpret_cast<uint4*>(&Ws[buf ^ 1][lrow[i] * LDT + lcol[i]]) = wn[i];
      }
    }
#endif
    __syncthreads();
    buf ^= 1;
  }

  for (int mt = 0; mt < 2; ++mt)
    for (int nt = 0; nt < 4; ++nt)
      for (int r = 0; r < 8; ++r) {
        int m = mb + wr * 32 + mt * 16 + r + 8 * hi;
        int n = nb + wc * 64 + nt * 16 + l15;
        C[(size_t)m * N + n] = acc[mt][nt][r];
      }
}

// ---------- kernel 3: RoPE + repack [B*S, NH*HD] f32 -> [B, NH, S, HD] bf16 ----------

__global__ void k_rope_repack(const float* __restrict__ in,
                              const float* __restrict__ cosT,
                              const float* __restrict__ sinT,
                              bf16* __restrict__ out, int NH, int applyRope) {
  int idx = blockIdx.x * blockDim.x + threadIdx.x;
  int total = B_ * S_ * NH * HD_;
  if (idx >= total) return;
  int d    = idx & (HD_ - 1);
  int rest = idx >> 7;
  int hh   = rest % NH;
  int m    = rest / NH;          // b*S + s
  int s    = m & (S_ - 1);
  int bb   = m >> 11;            // / S_
  float v = in[(size_t)m * (NH * HD_) + hh * HD_ + d];
  float r = v;
  if (applyRope) {
    float c  = cosT[s * HD_ + d];
    float sn = sinT[s * HD_ + d];
    float partner = in[(size_t)m * (NH * HD_) + hh * HD_ + ((d < 64) ? d + 64 : d - 64)];
    r = v * c + ((d < 64) ? -partner : partner) * sn;
  }
  out[(((size_t)(bb * NH + hh)) * S_ + s) * HD_ + d] = f2bf(r);
}

// ---------- kernel 4: causal flash attention ----------
// Block = 128 threads (4 waves). Block handles 64 query rows of one (b,h);
// each wave owns 16 rows. Key blocks of 64 staged in LDS (K via async DMA
// when available); V transposed through registers.

#define AT_Q  64
#define AT_KB 64
#define LDK   136   // 128 + 8 pad
#define LDV   72    // 64 + 8 pad
#define LDP   72

__global__ __launch_bounds__(128, 1)
void k_attention(const bf16* __restrict__ Q, const bf16* __restrict__ Kc,
                 const bf16* __restrict__ Vc, bf16* __restrict__ ctx) {
  __shared__ __align__(16) bf16 Kt[AT_KB * LDK];
  __shared__ __align__(16) bf16 Vt[HD_ * LDV];       // transposed: [d][key]
  __shared__ __align__(16) bf16 Pt[4 * 16 * LDP];    // per-wave P staging

  const int tid  = threadIdx.x;
  const int lane = tid & 31;
  const int w    = tid >> 5;
  const int hi   = (lane >> 4) & 1;
  const int l15  = lane & 15;

  const int qt  = blockIdx.x;
  const int bh  = blockIdx.y;
  const int h   = bh % H_;
  const int b   = bh / H_;
  const int kvh = h / GROUP_;

  const int qbase = qt * AT_Q;
  const int wrow  = qbase + w * 16;

  // resident Q fragments (A-layout), one per 32-wide k step of HD=128
  const bf16* qrowp = Q + (((size_t)(b * H_ + h)) * S_ + (wrow + l15)) * HD_;
  v16bf qf[4];
  for (int ks = 0; ks < 4; ++ks) {
    union { v16bf v; uint4 q[2]; } f;
    const bf16* p = qrowp + ks * 32 + 8 * hi;
    f.q[0] = *reinterpret_cast<const uint4*>(p);
    f.q[1] = *reinterpret_cast<const uint4*>(p + 16);
    qf[ks] = f.v;
  }

  v8f cacc[8] = {};
  float mi[8], li[8];
  for (int r = 0; r < 8; ++r) { mi[r] = -3.0e38f; li[r] = 0.f; }

  const bf16* Kbase = Kc + ((size_t)(b * KV_ + kvh)) * S_ * HD_;
  const bf16* Vbase = Vc + ((size_t)(b * KV_ + kvh)) * S_ * HD_;
  bf16* pW = &Pt[w * 16 * LDP];

  const int nkb = qt + 1;   // causal: keys only up to the diagonal block
  for (int kb = 0; kb < nkb; ++kb) {
    const int kbase = kb * AT_KB;

    // cooperative load: K row-major (async DMA if available), V transposed
    for (int i = 0; i < 8; ++i) {
      int c    = tid + i * 128;      // chunk 0..1023
      int key  = c >> 4;             // 16 chunks per key row
      int dcol = (c & 15) * 8;
#if HAVE_ASYNC_LDS
      async_copy16(&Kbase[(size_t)(kbase + key) * HD_ + dcol],
                   &Kt[key * LDK + dcol]);
#else
      uint4 kq = *reinterpret_cast<const uint4*>(
          &Kbase[(size_t)(kbase + key) * HD_ + dcol]);
      *reinterpret_cast<uint4*>(&Kt[key * LDK + dcol]) = kq;
#endif
      union { uint4 q; bf16 hh[8]; } vv;
      vv.q = *reinterpret_cast<const uint4*>(
          &Vbase[(size_t)(kbase + key) * HD_ + dcol]);
      for (int j = 0; j < 8; ++j) Vt[(dcol + j) * LDV + key] = vv.hh[j];
    }
#if HAVE_ASYNC_LDS
    async_wait0();
#endif
    __syncthreads();

    // scores: 4 key sub-tiles x 4 k-steps = 16 WMMA
    v8f sc[4];
    for (int nt = 0; nt < 4; ++nt) {
      v8f s = {};
      for (int ks = 0; ks < 4; ++ks) {
        v16bf kf = load_frag(Kt, nt * 16 + l15, LDK, ks * 32, lane);
        s = wmma_bf16(qf[ks], kf, s);
      }
      sc[nt] = s;
    }

    // scale + causal mask (matches reference's where(mask, -1e9, ...))
    for (int nt = 0; nt < 4; ++nt) {
      int key = kbase + nt * 16 + l15;
      for (int r = 0; r < 8; ++r) {
        int qr = wrow + r + 8 * hi;
        float v = sc[nt][r] * SCALE_;
        sc[nt][r] = (key > qr) ? -1.0e9f : v;
      }
    }

    // online softmax; row stats reduced across the 16 lanes of each half-wave
    float mnew[8], alpha[8];
    for (int r = 0; r < 8; ++r) {
      float rm = fmaxf(fmaxf(sc[0][r], sc[1][r]), fmaxf(sc[2][r], sc[3][r]));
      for (int m = 1; m < 16; m <<= 1) rm = fmaxf(rm, __shfl_xor(rm, m, 32));
      mnew[r]  = fmaxf(mi[r], rm);
      alpha[r] = __expf(mi[r] - mnew[r]);
      mi[r]    = mnew[r];
    }
    float rsum[8];
    for (int r = 0; r < 8; ++r) rsum[r] = 0.f;
    for (int nt = 0; nt < 4; ++nt)
      for (int r = 0; r < 8; ++r) {
        float p = __expf(sc[nt][r] - mnew[r]);
        sc[nt][r] = p;
        rsum[r] += p;
      }
    for (int r = 0; r < 8; ++r) {
      for (int m = 1; m < 16; m <<= 1) rsum[r] += __shfl_xor(rsum[r], m, 32);
      li[r] = li[r] * alpha[r] + rsum[r];
    }
    for (int i = 0; i < 8; ++i)
      for (int r = 0; r < 8; ++r) cacc[i][r] *= alpha[r];

    // re-layout P (C-layout -> A-layout) through per-wave LDS region;
    // same-wave DS ops are in order, so no barrier needed here.
    for (int nt = 0; nt < 4; ++nt)
      for (int r = 0; r < 8; ++r)
        pW[(r + 8 * hi) * LDP + nt * 16 + l15] = f2bf(sc[nt][r]);

    // ctx += P * V : 2 k-steps x 8 d sub-tiles = 16 WMMA
    for (int ks = 0; ks < 2; ++ks) {
      v16bf pf = load_frag(pW, l15, LDP, ks * 32, lane);
      for (int ntv = 0; ntv < 8; ++ntv) {
        v16bf vf = load_frag(Vt, ntv * 16 + l15, LDV, ks * 32, lane);
        cacc[ntv] = wmma_bf16(pf, vf, cacc[ntv]);
      }
    }
    __syncthreads();   // protect Kt/Vt before next key block
  }

  // epilogue: normalize and emit ctx as bf16 [B*S, H*HD] for the output GEMM
  for (int ntv = 0; ntv < 8; ++ntv)
    for (int r = 0; r < 8; ++r) {
      int srow = wrow + r + 8 * hi;
      int dcol = ntv * 16 + l15;
      ctx[((size_t)b * S_ + srow) * (H_ * HD_) + h * HD_ + dcol] =
          f2bf(cacc[ntv][r] / li[r]);
    }
}

// ---------- host ----------

extern "C" void kernel_launch(void* const* d_in, const int* in_sizes, int n_in,
                              void* d_out, int out_size, void* d_ws, size_t ws_size,
                              hipStream_t stream) {
  (void)in_sizes; (void)n_in; (void)out_size; (void)ws_size;
  const float* x    = (const float*)d_in[0];
  // d_in[1] = mask (causal, recomputed analytically — unused)
  const float* cosT = (const float*)d_in[2];
  const float* sinT = (const float*)d_in[3];
  const float* Wq   = (const float*)d_in[4];
  const float* Wk   = (const float*)d_in[5];
  const float* Wv   = (const float*)d_in[6];
  const float* Wo   = (const float*)d_in[7];
  float* out = (float*)d_out;

  char* ws = (char*)d_ws;
  size_t off = 0;
  auto alloc = [&](size_t bytes) -> char* {
    char* p = ws + off;
    off = (off + bytes + 255) & ~(size_t)255;
    return p;
  };
  const size_t M = (size_t)B_ * S_;              // 4096
  bf16*  x_bf  = (bf16*) alloc(M * DIN * 2);
  bf16*  Wq_bf = (bf16*) alloc((size_t)H_  * HD_ * DIN * 2);
  bf16*  Wk_bf = (bf16*) alloc((size_t)KV_ * HD_ * DIN * 2);
  bf16*  Wv_bf = (bf16*) alloc((size_t)KV_ * HD_ * DIN * 2);
  bf16*  Wo_bf = (bf16*) alloc((size_t)DIN * H_ * HD_ * 2);
  float* qf32  = (float*)alloc(M * H_  * HD_ * 4);
  float* kf32  = (float*)alloc(M * KV_ * HD_ * 4);
  float* vf32  = (float*)alloc(M * KV_ * HD_ * 4);
  bf16*  q_bf  = (bf16*) alloc(M * H_  * HD_ * 2);
  bf16*  k_bf  = (bf16*) alloc(M * KV_ * HD_ * 2);
  bf16*  v_bf  = (bf16*) alloc(M * KV_ * HD_ * 2);
  bf16*  c_bf  = (bf16*) alloc(M * H_  * HD_ * 2);

  auto cvt = [&](const float* src, bf16* dst, size_t n) {
    int blocks = (int)((n + 255) / 256);
    k_f32_to_bf16<<<blocks, 256, 0, stream>>>(src, dst, (int)n);
  };
  cvt(x,  x_bf,  M * DIN);
  cvt(Wq, Wq_bf, (size_t)H_  * HD_ * DIN);
  cvt(Wk, Wk_bf, (size_t)KV_ * HD_ * DIN);
  cvt(Wv, Wv_bf, (size_t)KV_ * HD_ * DIN);
  cvt(Wo, Wo_bf, (size_t)DIN * H_ * HD_);

  // QKV projections
  k_gemm_bf16<<<dim3((H_  * HD_) / GT_N, (unsigned)(M / GT_M)), 256, 0, stream>>>(
      x_bf, Wq_bf, qf32, (int)M, H_ * HD_, DIN);
  k_gemm_bf16<<<dim3((KV_ * HD_) / GT_N, (unsigned)(M / GT_M)), 256, 0, stream>>>(
      x_bf, Wk_bf, kf32, (int)M, KV_ * HD_, DIN);
  k_gemm_bf16<<<dim3((KV_ * HD_) / GT_N, (unsigned)(M / GT_M)), 256, 0, stream>>>(
      x_bf, Wv_bf, vf32, (int)M, KV_ * HD_, DIN);

  // RoPE + repack (v: convert/repack only)
  int tq = (int)(M * H_ * HD_);
  int tk = (int)(M * KV_ * HD_);
  k_rope_repack<<<(tq + 255) / 256, 256, 0, stream>>>(qf32, cosT, sinT, q_bf, H_, 1);
  k_rope_repack<<<(tk + 255) / 256, 256, 0, stream>>>(kf32, cosT, sinT, k_bf, KV_, 1);
  k_rope_repack<<<(tk + 255) / 256, 256, 0, stream>>>(vf32, cosT, sinT, v_bf, KV_, 0);

  // causal flash attention
  k_attention<<<dim3(S_ / AT_Q, B_ * H_), 128, 0, stream>>>(q_bf, k_bf, v_bf, c_bf);

  // output projection -> d_out (fp32)
  k_gemm_bf16<<<dim3(DIN / GT_N, (unsigned)(M / GT_M)), 256, 0, stream>>>(
      c_bf, Wo_bf, out, (int)M, DIN, DIN);
}
